// AttentionBlock_23364622090678
// MI455X (gfx1250) — matrix operand order repaired and checked by reading, not compiled, not run on hardware
//
#include <hip/hip_runtime.h>

// ---------------------------------------------------------------------------
// AttentionBlock for MI455X (gfx1250, wave32, WMMA + async-LDS staging)
//   B=4, C=256, H=W=64  ->  N=4096 tokens, head_dim=256, softmax scale 1/16
// Pipeline: groupnorm -> qkv gemm (wmma f16) -> flash attention (wmma f16,
// online softmax, Q in registers, K/V double-buffered in LDS via
// global_load_async_to_lds_b128) -> proj gemm (wmma f16) + residual.
// ---------------------------------------------------------------------------

typedef __attribute__((ext_vector_type(16))) _Float16 v16h;
typedef __attribute__((ext_vector_type(8)))  float    v8f;

#define B_   4
#define C_   256
#define N_   4096
#define GRP_ELEMS (32 * N_)   // elements per (batch, group)

// workspace offsets, in f16 elements
#define OFF_QKVW 0u
#define OFF_PROJW 196608u                 // 768*256
#define OFF_H    262144u                  // + 256*256
#define OFF_Q    (OFF_H  + 4194304u)      // B*N*C
#define OFF_K    (OFF_Q  + 4194304u)
#define OFF_VT   (OFF_K  + 4194304u)
#define OFF_AO   (OFF_VT + 4194304u)

union FragU { v16h v; uint4 u[2]; };

__device__ __forceinline__ v8f wmma16(v16h a, v16h b, v8f c) {
  // D = A(16x32 f16) * B(32x16 f16) + C(16x16 f32)
  return __builtin_amdgcn_wmma_f32_16x16x32_f16(false, a, false, b,
                                                (short)0, c, false, false);
}

// A fragment (16x32, f16), source row-major with leading dim ld (halves).
// ISA layout: lane L -> row M = L&15; halves 0..7 = K(8*(L>>4))..+7,
//             halves 8..15 = K(16+8*(L>>4))..+7.
__device__ __forceinline__ v16h load_a(const _Float16* base, int ld) {
  const int lane = threadIdx.x & 31;
  const _Float16* p = base + (lane & 15) * ld + ((lane >> 4) << 3);
  FragU f;
  f.u[0] = *(const uint4*)(p);
  f.u[1] = *(const uint4*)(p + 16);
  return f.v;
}

// B fragment (32x16, f16) where B[k][n] = src[n*ld + k] (src row n holds K
// contiguously). ISA layout: lane L -> col N = L&15; halves 0..7 = K(8*(L>>4))..+7,
//             halves 8..15 = K(16+8*(L>>4))..+7.
__device__ __forceinline__ v16h load_b(const _Float16* base, int ld) {
  const int lane = threadIdx.x & 31;
  const _Float16* p = base + (lane & 15) * ld + ((lane >> 4) << 3);
  FragU f;
  f.u[0] = *(const uint4*)(p);
  f.u[1] = *(const uint4*)(p + 16);
  return f.v;
}

// A-fragment from an LDS tile with ld = 32 halves.
__device__ __forceinline__ v16h load_a_lds(const _Float16* lp) {
  const int lane = threadIdx.x & 31;
  const _Float16* p = lp + (lane & 15) * 32 + ((lane >> 4) << 3);
  FragU f;
  f.u[0] = *(const uint4*)(p);
  f.u[1] = *(const uint4*)(p + 16);
  return f.v;
}

// CDNA5 async copy: 16 bytes global -> LDS per lane, tracked by ASYNCcnt.
__device__ __forceinline__ void async_b128(unsigned int lds_off, const _Float16* gp) {
  asm volatile("global_load_async_to_lds_b128 %0, %1, off"
               :: "v"(lds_off), "v"((unsigned long long)(size_t)gp)
               : "memory");
}
__device__ __forceinline__ void wait_async0() {
  asm volatile("s_wait_asynccnt 0x0" ::: "memory");
}

// ---------------------------------------------------------------------------
// 1) weight fp32 -> f16 repack
// ---------------------------------------------------------------------------
__global__ void __launch_bounds__(256)
convert_weights(const float* __restrict__ qkv_w, const float* __restrict__ proj_w,
                _Float16* __restrict__ wq, _Float16* __restrict__ wp) {
  int idx = blockIdx.x * 256 + threadIdx.x;
  if (idx < 196608) wq[idx] = (_Float16)qkv_w[idx];
  int j = idx - 196608;
  if (j >= 0 && j < 65536) wp[j] = (_Float16)proj_w[j];
}

// ---------------------------------------------------------------------------
// 2) GroupNorm: one block per (batch, group); writes h token-major f16
// ---------------------------------------------------------------------------
__global__ void __launch_bounds__(256)
group_norm(const float* __restrict__ x, const float* __restrict__ gn_w,
           const float* __restrict__ gn_b, _Float16* __restrict__ h) {
  const int b = blockIdx.x >> 3;
  const int g = blockIdx.x & 7;
  const float* xb = x + (size_t)(b * C_ + g * 32) * N_;

  float s = 0.f, s2 = 0.f;
  for (int i = threadIdx.x; i < GRP_ELEMS; i += 256) {
    float v = xb[i];
    s += v; s2 += v * v;
  }
  #pragma unroll
  for (int m = 1; m <= 16; m <<= 1) {
    s  += __shfl_xor(s,  m, 32);
    s2 += __shfl_xor(s2, m, 32);
  }
  __shared__ float red[16];
  const int wave = threadIdx.x >> 5, lane = threadIdx.x & 31;
  if (lane == 0) { red[wave * 2] = s; red[wave * 2 + 1] = s2; }
  __syncthreads();
  if (threadIdx.x == 0) {
    float ts = 0.f, t2 = 0.f;
    #pragma unroll
    for (int w = 0; w < 8; ++w) { ts += red[2 * w]; t2 += red[2 * w + 1]; }
    float mean = ts * (1.0f / GRP_ELEMS);
    float var  = t2 * (1.0f / GRP_ELEMS) - mean * mean;
    red[0] = mean;
    red[1] = rsqrtf(var + 1e-5f);
  }
  __syncthreads();
  const float mean = red[0], rstd = red[1];

  for (int i = threadIdx.x; i < GRP_ELEMS; i += 256) {
    int c  = i >> 12;
    int n  = i & (N_ - 1);
    int ch = g * 32 + c;
    float v = (xb[i] - mean) * rstd * gn_w[ch] + gn_b[ch];
    h[((size_t)(b * N_ + n)) * C_ + ch] = (_Float16)v;
  }
}

// ---------------------------------------------------------------------------
// 3) QKV GEMM: grid (12, 64, 4): x = 64-output block, y = 64-token, z = batch
// ---------------------------------------------------------------------------
__global__ void __launch_bounds__(128)
qkv_gemm(const _Float16* __restrict__ h, const _Float16* __restrict__ wq,
         const float* __restrict__ qkv_b, _Float16* __restrict__ q,
         _Float16* __restrict__ kk, _Float16* __restrict__ vT) {
  const int b = blockIdx.z, tblk = blockIdx.y, oblk = blockIdx.x;
  const int wave = threadIdx.x >> 5, lane = threadIdx.x & 31;
  const int mtok = tblk * 64 + wave * 16;
  const _Float16* A = h + ((size_t)(b * N_ + mtok)) * C_;

  v8f acc[4];
  #pragma unroll
  for (int t = 0; t < 4; ++t) acc[t] = (v8f){0,0,0,0,0,0,0,0};

  for (int k0 = 0; k0 < C_; k0 += 32) {
    v16h a = load_a(A + k0, C_);
    #pragma unroll
    for (int nt = 0; nt < 4; ++nt) {
      const int o0 = oblk * 64 + nt * 16;
      v16h bb = load_b(wq + (size_t)o0 * C_ + k0, C_);
      acc[nt] = wmma16(a, bb, acc[nt]);
    }
  }

  const int ncol = lane & 15, rbase = (lane >> 4) << 3;
  #pragma unroll
  for (int nt = 0; nt < 4; ++nt) {
    const int o = oblk * 64 + nt * 16 + ncol;
    const float bias = qkv_b[o];
    #pragma unroll
    for (int r = 0; r < 8; ++r) {
      const int t = mtok + rbase + r;
      float val = acc[nt][r] + bias;
      if (o < 256) {            // Q: fold softmax scale 1/sqrt(256)
        q[((size_t)(b * N_ + t)) * C_ + o] = (_Float16)(val * 0.0625f);
      } else if (o < 512) {     // K
        kk[((size_t)(b * N_ + t)) * C_ + (o - 256)] = (_Float16)val;
      } else {                  // V transposed [c][token]
        vT[((size_t)(b * C_ + (o - 512))) * N_ + t] = (_Float16)val;
      }
    }
  }
}

// ---------------------------------------------------------------------------
// 4) Flash attention: grid (64, 4) = (64-query block, batch), 256 threads.
//    Q in registers; K/V tiles double-buffered in LDS via async-to-LDS.
// ---------------------------------------------------------------------------
__global__ void __launch_bounds__(256)
flash_attn(const _Float16* __restrict__ q, const _Float16* __restrict__ k,
           const _Float16* __restrict__ vT, _Float16* __restrict__ ao) {
  __shared__ __align__(16) _Float16 ldsK[2][32 * 256];   // 2 x 16 KB
  __shared__ __align__(16) _Float16 ldsV[2][256 * 32];   // 2 x 16 KB
  __shared__ __align__(16) _Float16 ldsP[8][16][32];     // 8 KB, wave-private

  const int b = blockIdx.y, qblk = blockIdx.x;
  const int wave = threadIdx.x >> 5, lane = threadIdx.x & 31;
  const int mt = wave >> 1;          // 16-query tile (0..3)
  const int ch = wave & 1;           // 128-channel half
  const int tok0 = qblk * 64 + mt * 16;

  const _Float16* qb  = q  + ((size_t)(b * N_ + tok0)) * C_;
  const _Float16* kb  = k  + (size_t)b * N_ * C_;
  const _Float16* vtb = vT + (size_t)b * C_ * N_;

  // ---- preload Q fragments (16 x 256) into registers: reused 128 times
  v16h qf[8];
  #pragma unroll
  for (int kk0 = 0; kk0 < 8; ++kk0) qf[kk0] = load_a(qb + kk0 * 32, C_);

  // cooperative async fill of one K/V tile set (32 keys): 8 b128 per thread
  const unsigned int tid = threadIdx.x;
  auto issue = [&](int buf, int kv) {
    unsigned int kBase = (unsigned int)(size_t)(void*)&ldsK[buf][0];
    unsigned int vBase = (unsigned int)(size_t)(void*)&ldsV[buf][0];
    #pragma unroll
    for (int j = 0; j < 4; ++j) {          // K tile: 32 rows x 256 ch
      int ckk = j * 256 + (int)tid;        // 1024 chunks of 8 halves
      int row = ckk >> 5, off = (ckk & 31) * 8;
      async_b128(kBase + (unsigned int)(row * 256 + off) * 2u,
                 kb + (size_t)(kv + row) * C_ + off);
    }
    #pragma unroll
    for (int j = 0; j < 4; ++j) {          // V tile: 256 ch x 32 keys
      int cv = j * 256 + (int)tid;
      int c = cv >> 2, off = (cv & 3) * 8;
      async_b128(vBase + (unsigned int)(c * 32 + off) * 2u,
                 vtb + (size_t)c * N_ + kv + off);
    }
  };

  v8f acc[8];
  #pragma unroll
  for (int t = 0; t < 8; ++t) acc[t] = (v8f){0,0,0,0,0,0,0,0};
  float rowm[8], rowl[8];
  #pragma unroll
  for (int r = 0; r < 8; ++r) { rowm[r] = -3.0e38f; rowl[r] = 0.f; }

  const int ncol = lane & 15, rbase = (lane >> 4) << 3;
  _Float16* lp = &ldsP[wave][0][0];

  issue(0, 0);                              // prologue fill

  for (int i = 0; i < 128; ++i) {
    const int kv  = i * 32;
    const int cur = i & 1;
    wait_async0();                          // my fills of buf[cur] complete
    __syncthreads();                        // everyone's fills visible; prev
                                            // readers of buf[cur^1] are past
    if (i + 1 < 128) issue(cur ^ 1, kv + 32);   // overlap next tile with math

    const _Float16* Kt = &ldsK[cur][0];
    const _Float16* Vt = &ldsV[cur][0];

    // ---- S = Q * K^T for keys [kv, kv+32) : two 16x16 tiles
    v8f s0 = (v8f){0,0,0,0,0,0,0,0};
    v8f s1 = (v8f){0,0,0,0,0,0,0,0};
    #pragma unroll
    for (int kk0 = 0; kk0 < 8; ++kk0) {
      v16h b0 = load_b(Kt + kk0 * 32, C_);
      v16h b1 = load_b(Kt + 16 * C_ + kk0 * 32, C_);
      s0 = wmma16(qf[kk0], b0, s0);
      s1 = wmma16(qf[kk0], b1, s1);
    }

    // ---- online softmax (C-layout: a row's 16 cols live in one 16-lane half)
    #pragma unroll
    for (int r = 0; r < 8; ++r) {
      float pm = fmaxf(s0[r], s1[r]);
      pm = fmaxf(pm, __shfl_xor(pm, 1, 32));
      pm = fmaxf(pm, __shfl_xor(pm, 2, 32));
      pm = fmaxf(pm, __shfl_xor(pm, 4, 32));
      pm = fmaxf(pm, __shfl_xor(pm, 8, 32));
      float mnew = fmaxf(rowm[r], pm);
      float corr = __expf(rowm[r] - mnew);
      rowm[r] = mnew;
      float e0 = __expf(s0[r] - mnew);
      float e1 = __expf(s1[r] - mnew);
      s0[r] = e0; s1[r] = e1;
      float ps = e0 + e1;
      ps += __shfl_xor(ps, 1, 32);
      ps += __shfl_xor(ps, 2, 32);
      ps += __shfl_xor(ps, 4, 32);
      ps += __shfl_xor(ps, 8, 32);
      rowl[r] = rowl[r] * corr + ps;
      #pragma unroll
      for (int t = 0; t < 8; ++t) acc[t][r] *= corr;
    }

    // ---- stage P (C-layout) to wave-private LDS, reload as A-fragment
    #pragma unroll
    for (int r = 0; r < 8; ++r) {
      const int row = rbase + r;
      lp[row * 32 + ncol]      = (_Float16)s0[r];
      lp[row * 32 + 16 + ncol] = (_Float16)s1[r];
    }
    v16h pa = load_a_lds(lp);

    // ---- O += P * V (V tile in LDS, transposed: ld = 32 keys)
    #pragma unroll
    for (int t = 0; t < 8; ++t) {
      const int cb = ch * 128 + t * 16;
      v16h bv = load_b(Vt + cb * 32, 32);
      acc[t] = wmma16(pa, bv, acc[t]);
    }
  }

  // ---- normalize and emit attention output, token-major f16
  #pragma unroll
  for (int r = 0; r < 8; ++r) rowl[r] = 1.0f / rowl[r];
  #pragma unroll
  for (int t = 0; t < 8; ++t) {
    const int chn = ch * 128 + t * 16 + ncol;
    #pragma unroll
    for (int r = 0; r < 8; ++r) {
      const int tok = tok0 + rbase + r;
      ao[((size_t)(b * N_ + tok)) * C_ + chn] = (_Float16)(acc[t][r] * rowl[r]);
    }
  }
}

// ---------------------------------------------------------------------------
// 5) Projection GEMM + bias + residual, back to [B, C, H, W] fp32
//    grid (64, 4, 4): x = 64-token block, y = 64-channel block, z = batch
// ---------------------------------------------------------------------------
__global__ void __launch_bounds__(128)
proj_gemm(const _Float16* __restrict__ ao, const _Float16* __restrict__ wp,
          const float* __restrict__ proj_b, const float* __restrict__ x,
          float* __restrict__ out) {
  const int b = blockIdx.z, cblk = blockIdx.y, nblk = blockIdx.x;
  const int wave = threadIdx.x >> 5, lane = threadIdx.x & 31;
  const int crow0 = cblk * 64 + wave * 16;
  const _Float16* A = wp + (size_t)crow0 * C_;                 // proj_w[c][o]
  const _Float16* Bb = ao + ((size_t)(b * N_ + nblk * 64)) * C_;

  v8f acc[4];
  #pragma unroll
  for (int t = 0; t < 4; ++t) acc[t] = (v8f){0,0,0,0,0,0,0,0};

  for (int k0 = 0; k0 < C_; k0 += 32) {
    v16h a = load_a(A + k0, C_);
    #pragma unroll
    for (int nt = 0; nt < 4; ++nt) {
      v16h bb = load_b(Bb + (size_t)(nt * 16) * C_ + k0, C_);
      acc[nt] = wmma16(a, bb, acc[nt]);
    }
  }

  const int ncol = lane & 15, rbase = (lane >> 4) << 3;
  #pragma unroll
  for (int nt = 0; nt < 4; ++nt) {
    const int tok = nblk * 64 + nt * 16 + ncol;
    #pragma unroll
    for (int r = 0; r < 8; ++r) {
      const int c = crow0 + rbase + r;
      const size_t idx = ((size_t)(b * C_ + c)) * N_ + tok;
      out[idx] = acc[nt][r] + proj_b[c] + x[idx];
    }
  }
}

// ---------------------------------------------------------------------------
extern "C" void kernel_launch(void* const* d_in, const int* in_sizes, int n_in,
                              void* d_out, int out_size, void* d_ws, size_t ws_size,
                              hipStream_t stream) {
  const float* x      = (const float*)d_in[0];
  const float* gn_w   = (const float*)d_in[1];
  const float* gn_b   = (const float*)d_in[2];
  const float* qkv_w  = (const float*)d_in[3];
  const float* qkv_b  = (const float*)d_in[4];
  const float* proj_w = (const float*)d_in[5];
  const float* proj_b = (const float*)d_in[6];
  float* out = (float*)d_out;
  _Float16* ws = (_Float16*)d_ws;

  convert_weights<<<1024, 256, 0, stream>>>(qkv_w, proj_w,
                                            ws + OFF_QKVW, ws + OFF_PROJW);
  group_norm<<<32, 256, 0, stream>>>(x, gn_w, gn_b, ws + OFF_H);
  qkv_gemm<<<dim3(12, 64, 4), 128, 0, stream>>>(ws + OFF_H, ws + OFF_QKVW, qkv_b,
                                                ws + OFF_Q, ws + OFF_K, ws + OFF_VT);
  flash_attn<<<dim3(64, 4), 256, 0, stream>>>(ws + OFF_Q, ws + OFF_K,
                                              ws + OFF_VT, ws + OFF_AO);
  proj_gemm<<<dim3(64, 4, 4), 128, 0, stream>>>(ws + OFF_AO, ws + OFF_PROJW,
                                                proj_b, x, out);
}
